// AitchisonMultiHeadAttention_58317065945691
// MI455X (gfx1250) — compile-verified, matrix operand order checked by reading, not run on hardware
//
#include <hip/hip_runtime.h>
#include <hip/hip_bf16.h>

#define EMBED 1024
#define NHEAD 16
#define HDIM  64
#define BSZ   2
#define TQS   2048
#define TKS   2048

typedef __attribute__((ext_vector_type(16))) __bf16 v16bf;
typedef __attribute__((ext_vector_type(8)))  __bf16 v8bf;
typedef __attribute__((ext_vector_type(8)))  float  v8f;

// ---------------------------------------------------------------------------
// Fragment loader for V_WMMA_F32_16X16X32_BF16 A/B operands.
// ISA 7.12.2 (16-bit A-matrix 16x32): lanes 0-15 hold K={0..7,16..23},
// lanes 16-31 hold K={8..15,24..31}; 2 elements packed per VGPR in K order.
// base points at (row, k_tile_base); rows are K-contiguous in memory.
// ---------------------------------------------------------------------------
__device__ inline v16bf load_frag(const __bf16* base, int lane) {
    int koff = (lane & 16) ? 8 : 0;
    v8bf lo = *(const v8bf*)(base + koff);
    v8bf hi = *(const v8bf*)(base + 16 + koff);
    return __builtin_shufflevector(lo, hi, 0, 1, 2, 3, 4, 5, 6, 7,
                                           8, 9, 10, 11, 12, 13, 14, 15);
}

// ---------------------------------------------------------------------------
// fp32 -> bf16 conversion pass
// ---------------------------------------------------------------------------
__global__ void cvt_f32_bf16(const float* __restrict__ in,
                             __bf16* __restrict__ out, int n) {
    int i = blockIdx.x * blockDim.x + threadIdx.x;
    if (i < n) out[i] = (__bf16)in[i];
}

// ---------------------------------------------------------------------------
// Projection GEMM: C[M,N] = A[M,K] @ W[N,K]^T + bias  (K = N = 1024)
// Wave computes a 16x64 strip (one full head in N).
// MODE 0: write fp32 [M, EMBED]            (final output projection)
// MODE 1: head-center + clr bias + scale, write bf16 [B,H,T,D]    (q, k)
// MODE 2: write bf16 TRANSPOSED [B,H,D,T]                          (v)
// Block: 128 threads = 4 waves stacked along M. Grid: (M/64, N/64).
// ---------------------------------------------------------------------------
template <int MODE>
__global__ __launch_bounds__(128) void gemm_proj(
    const __bf16* __restrict__ A, const __bf16* __restrict__ W,
    const float* __restrict__ bias, const float* __restrict__ clr,
    float scale, void* __restrict__ out, int M) {
    const int lane = threadIdx.x & 31;
    const int wave = threadIdx.x >> 5;
    const int row0 = blockIdx.x * 64 + wave * 16;
    const int col0 = blockIdx.y * 64;
    const int ln15 = lane & 15;
    const int half = (lane & 16) ? 8 : 0;

    v8f acc[4] = {};
    const __bf16* arow = A + (size_t)(row0 + ln15) * EMBED;

    for (int k = 0; k < EMBED; k += 32) {
        __builtin_prefetch(arow + k + 128, 0, 0);
        v16bf af = load_frag(arow + k, lane);
#pragma unroll
        for (int j = 0; j < 4; ++j) {
            const __bf16* wrow = W + (size_t)(col0 + j * 16 + ln15) * EMBED;
            v16bf bf = load_frag(wrow + k, lane);
            acc[j] = __builtin_amdgcn_wmma_f32_16x16x32_bf16(
                false, af, false, bf, (short)0, acc[j], false, false);
        }
    }

    // add projection bias (must happen before centering: ref centers y+b)
#pragma unroll
    for (int j = 0; j < 4; ++j) {
        float bv = bias[col0 + j * 16 + ln15];
#pragma unroll
        for (int r = 0; r < 8; ++r) acc[j][r] += bv;
    }

    if (MODE == 0) {
        float* O = (float*)out;
#pragma unroll
        for (int r = 0; r < 8; ++r) {
            int row = row0 + r + half;
#pragma unroll
            for (int j = 0; j < 4; ++j)
                O[(size_t)row * EMBED + col0 + j * 16 + ln15] = acc[j][r];
        }
    } else {
        const int h = blockIdx.y;  // N strip == head
        __bf16* O = (__bf16*)out;
#pragma unroll
        for (int r = 0; r < 8; ++r) {
            float mean = 0.0f;
            if (MODE == 1) {
                // row sum over the 64 head columns: 4 accs x 16 lanes
                float s = acc[0][r] + acc[1][r] + acc[2][r] + acc[3][r];
                s += __shfl_xor(s, 1);
                s += __shfl_xor(s, 2);
                s += __shfl_xor(s, 4);
                s += __shfl_xor(s, 8);
                mean = s * (1.0f / 64.0f);
            }
            int row = row0 + r + half;
            int b = row / TQS;
            int t = row % TQS;
#pragma unroll
            for (int j = 0; j < 4; ++j) {
                int d = j * 16 + ln15;
                float v = acc[j][r] - mean;
                if (MODE == 1) {
                    v = (v + clr[h * HDIM + d]) * scale;
                    size_t base = (((size_t)b * NHEAD + h) * TQS + t) * HDIM;
                    O[base + d] = (__bf16)v;
                } else {
                    // transposed [B,H,D,T] so attention reads K-contiguous
                    size_t base = (((size_t)b * NHEAD + h) * HDIM + d) * TKS;
                    O[base + t] = (__bf16)v;
                }
            }
        }
    }
}

// ---------------------------------------------------------------------------
// Flash attention: per (b, h), each wave independently owns a 32-row q tile
// (two 16-row WMMA tiles sharing all K/V fragments), marching over keys in
// 64-wide chunks so each online-softmax update + acc rescale amortizes over
// 32 WMMAs. K/V fragments load straight from global (L2/WGP$ serve the
// cross-wave reuse); the only LDS traffic is the per-wave P re-layout buffer
// (no barriers anywhere).
// Qc/Kc: pre-centered (+clr bias, q pre-scaled by 1/sqrt(D)) bf16 [B,H,T,D].
// Vt: bf16 [B,H,D,T] (transposed). Output: bf16 [B,T,EMBED].
// ---------------------------------------------------------------------------
__global__ __launch_bounds__(256) void attn_flash(
    const __bf16* __restrict__ Qc, const __bf16* __restrict__ Kc,
    const __bf16* __restrict__ Vt, const unsigned char* __restrict__ kmask,
    __bf16* __restrict__ O) {
    __shared__ alignas(16) __bf16 Pt[8][2][16 * 64];  // 32 KB

    const int lane = threadIdx.x & 31;
    const int wave = threadIdx.x >> 5;
    const int ln15 = lane & 15;
    const int half = (lane & 16) ? 8 : 0;
    const int b = blockIdx.z;
    const int h = blockIdx.y;
    const int q0 = (blockIdx.x * 8 + wave) * 32;

    const __bf16* Qbh = Qc + ((size_t)b * NHEAD + h) * TQS * HDIM;
    const __bf16* Kbh = Kc + ((size_t)b * NHEAD + h) * TKS * HDIM;
    const __bf16* Vbh = Vt + ((size_t)b * NHEAD + h) * HDIM * TKS;

    // Q operand fragments: 2 row tiles x 2 K-halves, resident for whole loop
    v16bf aq[2][2];
#pragma unroll
    for (int rt = 0; rt < 2; ++rt) {
        const __bf16* qrow = Qbh + (size_t)(q0 + rt * 16 + ln15) * HDIM;
        aq[rt][0] = load_frag(qrow + 0, lane);
        aq[rt][1] = load_frag(qrow + 32, lane);
    }

    float mrun[2][8], lrun[2][8];
#pragma unroll
    for (int rt = 0; rt < 2; ++rt)
#pragma unroll
        for (int r = 0; r < 8; ++r) {
            mrun[rt][r] = -__builtin_inff();
            lrun[rt][r] = 0.0f;
        }
    v8f acc[2][4] = {};

    for (int kb = 0; kb < TKS; kb += 64) {
        // K fragments (B operand: d x kcol) for 4 column subtiles,
        // shared by both row tiles; mask becomes a {0,1} factor per column.
        v16bf bk[4][2];
        float mfac[4];
#pragma unroll
        for (int nsub = 0; nsub < 4; ++nsub) {
            int kcg = kb + nsub * 16 + ln15;
            const __bf16* krow = Kbh + (size_t)kcg * HDIM;
            __builtin_prefetch(krow + 64 * HDIM, 0, 0);
            bk[nsub][0] = load_frag(krow + 0, lane);
            bk[nsub][1] = load_frag(krow + 32, lane);
            mfac[nsub] = kmask[(size_t)b * TKS + kcg] ? 0.0f : 1.0f;
        }

        v16bf pf[2][2];
#pragma unroll
        for (int rt = 0; rt < 2; ++rt) {
            // S = q_c @ k_c^T : four 16x16 column subtiles, K=64 = 2 WMMA each
            v8f s[4];
#pragma unroll
            for (int nsub = 0; nsub < 4; ++nsub) {
                v8f z = {};
                z = __builtin_amdgcn_wmma_f32_16x16x32_bf16(
                    false, aq[rt][0], false, bk[nsub][0], (short)0, z, false, false);
                z = __builtin_amdgcn_wmma_f32_16x16x32_bf16(
                    false, aq[rt][1], false, bk[nsub][1], (short)0, z, false, false);
                s[nsub] = z;
            }

            // online softmax: one update covers all 64 key columns.
            // Row max may include masked columns (over-estimate is fine:
            // flash softmax is invariant to the reference point); masked
            // probabilities are zeroed multiplicatively before l / P@V.
            float alpha[8];
#pragma unroll
            for (int r = 0; r < 8; ++r) {
                float mx = fmaxf(fmaxf(s[0][r], s[1][r]),
                                 fmaxf(s[2][r], s[3][r]));
                mx = fmaxf(mx, __shfl_xor(mx, 1));
                mx = fmaxf(mx, __shfl_xor(mx, 2));
                mx = fmaxf(mx, __shfl_xor(mx, 4));
                mx = fmaxf(mx, __shfl_xor(mx, 8));
                float mnew = fmaxf(mrun[rt][r], mx);
                float a = __expf(mrun[rt][r] - mnew);
                float p0 = __expf(s[0][r] - mnew) * mfac[0];
                float p1 = __expf(s[1][r] - mnew) * mfac[1];
                float p2 = __expf(s[2][r] - mnew) * mfac[2];
                float p3 = __expf(s[3][r] - mnew) * mfac[3];
                float ps = (p0 + p1) + (p2 + p3);
                ps += __shfl_xor(ps, 1);
                ps += __shfl_xor(ps, 2);
                ps += __shfl_xor(ps, 4);
                ps += __shfl_xor(ps, 8);
                lrun[rt][r] = lrun[rt][r] * a + ps;
                mrun[rt][r] = mnew;
                alpha[r] = a;
                __bf16* prow = &Pt[wave][rt][(r + half) * 64];
                prow[ln15] = (__bf16)p0;
                prow[16 + ln15] = (__bf16)p1;
                prow[32 + ln15] = (__bf16)p2;
                prow[48 + ln15] = (__bf16)p3;
            }

            // reload P (16x64) as two A fragments (per-wave LDS, in-order DS)
            pf[rt][0] = load_frag(&Pt[wave][rt][ln15 * 64], lane);
            pf[rt][1] = load_frag(&Pt[wave][rt][ln15 * 64 + 32], lane);

            // rescale running accumulators once per 64 columns
#pragma unroll
            for (int j = 0; j < 4; ++j)
#pragma unroll
                for (int r = 0; r < 8; ++r) acc[rt][j][r] *= alpha[r];
        }

        // O += P @ V : V^T [d][t] makes each B fragment two b128 loads
#pragma unroll
        for (int j = 0; j < 4; ++j) {
            int d = j * 16 + ln15;
            const __bf16* vrow = Vbh + (size_t)d * TKS + kb;
            __builtin_prefetch(vrow + 64, 0, 0);
            v16bf bv0 = load_frag(vrow, lane);
            v16bf bv1 = load_frag(vrow + 32, lane);
#pragma unroll
            for (int rt = 0; rt < 2; ++rt) {
                acc[rt][j] = __builtin_amdgcn_wmma_f32_16x16x32_bf16(
                    false, pf[rt][0], false, bv0, (short)0, acc[rt][j], false, false);
                acc[rt][j] = __builtin_amdgcn_wmma_f32_16x16x32_bf16(
                    false, pf[rt][1], false, bv1, (short)0, acc[rt][j], false, false);
            }
        }
    }

    // epilogue: normalize and write merged heads [B, T, EMBED] bf16
#pragma unroll
    for (int rt = 0; rt < 2; ++rt)
#pragma unroll
        for (int r = 0; r < 8; ++r) {
            float inv = 1.0f / lrun[rt][r];
            int row = q0 + rt * 16 + r + half;
#pragma unroll
            for (int j = 0; j < 4; ++j) {
                int e = h * HDIM + j * 16 + ln15;
                O[((size_t)b * TQS + row) * EMBED + e] =
                    (__bf16)(acc[rt][j][r] * inv);
            }
        }
}

// ---------------------------------------------------------------------------
// Host-side orchestration
// ---------------------------------------------------------------------------
extern "C" void kernel_launch(void* const* d_in, const int* in_sizes, int n_in,
                              void* d_out, int out_size, void* d_ws,
                              size_t ws_size, hipStream_t stream) {
    (void)in_sizes; (void)n_in; (void)out_size; (void)ws_size;
    const float* query = (const float*)d_in[0];
    const float* key   = (const float*)d_in[1];
    const float* value = (const float*)d_in[2];
    const unsigned char* kmask = (const unsigned char*)d_in[3];
    const float* Wq = (const float*)d_in[4];
    const float* bq = (const float*)d_in[5];
    const float* Wk = (const float*)d_in[6];
    const float* bk = (const float*)d_in[7];
    const float* Wv = (const float*)d_in[8];
    const float* bv = (const float*)d_in[9];
    const float* Wo = (const float*)d_in[10];
    const float* bo = (const float*)d_in[11];
    const float* clr_q = (const float*)d_in[12];
    const float* clr_k = (const float*)d_in[13];

    const int M = BSZ * TQS;                 // 4096 rows
    const int n_act = M * EMBED;             // 4,194,304
    const int n_w = EMBED * EMBED;           // 1,048,576

    char* ws = (char*)d_ws;
    size_t off = 0;
    __bf16* qb  = (__bf16*)(ws + off); off += (size_t)n_act * 2;   // 8 MB
    __bf16* kb  = (__bf16*)(ws + off); off += (size_t)n_act * 2;   // 8 MB
    __bf16* vb  = (__bf16*)(ws + off); off += (size_t)n_act * 2;   // 8 MB
    __bf16* wqb = (__bf16*)(ws + off); off += (size_t)n_w * 2;     // 2 MB
    __bf16* wkb = (__bf16*)(ws + off); off += (size_t)n_w * 2;
    __bf16* wvb = (__bf16*)(ws + off); off += (size_t)n_w * 2;
    __bf16* wob = (__bf16*)(ws + off); off += (size_t)n_w * 2;
    __bf16* Qc  = (__bf16*)(ws + off); off += (size_t)n_act * 2;   // [B,H,T,D]
    __bf16* Kc  = (__bf16*)(ws + off); off += (size_t)n_act * 2;   // [B,H,T,D]
    __bf16* Vtp = (__bf16*)(ws + off); off += (size_t)n_act * 2;   // [B,H,D,T]
    __bf16* Ob  = (__bf16*)(ws + off); off += (size_t)n_act * 2;   // [B,T,E]

    const int CT = 256;
    cvt_f32_bf16<<<(n_act + CT - 1) / CT, CT, 0, stream>>>(query, qb, n_act);
    cvt_f32_bf16<<<(n_act + CT - 1) / CT, CT, 0, stream>>>(key, kb, n_act);
    cvt_f32_bf16<<<(n_act + CT - 1) / CT, CT, 0, stream>>>(value, vb, n_act);
    cvt_f32_bf16<<<(n_w + CT - 1) / CT, CT, 0, stream>>>(Wq, wqb, n_w);
    cvt_f32_bf16<<<(n_w + CT - 1) / CT, CT, 0, stream>>>(Wk, wkb, n_w);
    cvt_f32_bf16<<<(n_w + CT - 1) / CT, CT, 0, stream>>>(Wv, wvb, n_w);
    cvt_f32_bf16<<<(n_w + CT - 1) / CT, CT, 0, stream>>>(Wo, wob, n_w);

    dim3 ggrid(M / 64, EMBED / 64);          // (64, 16)
    const float inv_scale = 0.125f;          // 1 / sqrt(HDIM)
    gemm_proj<1><<<ggrid, 128, 0, stream>>>(qb, wqb, bq, clr_q, inv_scale, Qc, M);
    gemm_proj<1><<<ggrid, 128, 0, stream>>>(kb, wkb, bk, clr_k, 1.0f, Kc, M);
    gemm_proj<2><<<ggrid, 128, 0, stream>>>(vb, wvb, bv, nullptr, 1.0f, Vtp, M);

    // each wave: 32 q rows; 8 waves/block -> 256 q rows per block
    attn_flash<<<dim3(TQS / 256, NHEAD, BSZ), 256, 0, stream>>>(Qc, Kc, Vtp,
                                                                kmask, Ob);

    gemm_proj<0><<<ggrid, 128, 0, stream>>>(Ob, wob, bo, nullptr, 1.0f,
                                            (float*)d_out, M);
}
